// GINConvNet_33904471834794
// MI455X (gfx1250) — matrix operand order repaired and checked by reading, not compiled, not run on hardware
//
#include <hip/hip_runtime.h>
#include <hip/hip_bf16.h>

typedef __attribute__((ext_vector_type(16))) _Float16 v16h;
typedef __attribute__((ext_vector_type(8)))  _Float16 v8h;
typedef __attribute__((ext_vector_type(8)))  float    v8f;

#define N_NODES 200000
#define N_EDGES 2000000
#define NB      1024
#define SEQ     1000
#define EMB     128
#define VOC     26
#define LOUT    121         // 128-8+1
#define CONV_K  3872        // 32*121 == 121 k-steps of 32
#define SLD     228         // padded LDS stride for S (conflict-free)
#define TLD     33          // padded LDS stride for GIN transpose buffer

static __device__ __forceinline__ v8f zero8() {
  v8f z;
#pragma unroll
  for (int i = 0; i < 8; ++i) z[i] = 0.f;
  return z;
}

static __device__ __forceinline__ v8f wmma16(v16h a, v16h b, v8f c) {
  return __builtin_amdgcn_wmma_f32_16x16x32_f16(false, a, false, b, (short)0, c, false, false);
}

// B fragment from pre-transposed f16 weights Bt[N][ldb]; K = kb + (hi?8:0) + {0..7, 16..23}
static __device__ __forceinline__ v16h load_bT(const _Float16* __restrict__ Bt, int ldb,
                                               int nbase, int kb, int lane) {
  int n  = nbase + (lane & 15);
  int ko = kb + ((lane & 16) ? 8 : 0);
  const _Float16* p = Bt + (size_t)n * ldb + ko;
  v8h lo = *(const v8h*)p;
  v8h hi = *(const v8h*)(p + 16);
  v16h r;
#pragma unroll
  for (int i = 0; i < 8; ++i) { r[i] = lo[i]; r[8 + i] = hi[i]; }
  return r;
}

// A fragment from f32 row pointer (row of >=32 valid floats starting at kb)
static __device__ __forceinline__ v16h load_aF32(const float* __restrict__ row, int kb, int lane) {
  const float* p = row + kb + ((lane & 16) ? 8 : 0);
  v16h r;
#pragma unroll
  for (int i = 0; i < 8; ++i) { r[i] = (_Float16)p[i]; r[8 + i] = (_Float16)p[16 + i]; }
  return r;
}

// A fragment from f16 matrix A[M][lda]
static __device__ __forceinline__ v16h load_aF16(const _Float16* __restrict__ A, int lda,
                                                 int mbase, int kb, int lane) {
  const _Float16* p = A + (size_t)(mbase + (lane & 15)) * lda + kb + ((lane & 16) ? 8 : 0);
  v8h lo = *(const v8h*)p;
  v8h hi = *(const v8h*)(p + 16);
  v16h r;
#pragma unroll
  for (int i = 0; i < 8; ++i) { r[i] = lo[i]; r[8 + i] = hi[i]; }
  return r;
}

// ---------------- weight prep ----------------
__global__ void k_transpose_f16(const float* __restrict__ B, _Float16* __restrict__ Bt,
                                int K, int N, int Kpad) {
  int idx = blockIdx.x * blockDim.x + threadIdx.x;
  if (idx >= N * Kpad) return;
  int n = idx / Kpad, k = idx % Kpad;
  Bt[idx] = (k < K) ? (_Float16)B[(size_t)k * N + n] : (_Float16)0.f;
}

// Ebt[l][v*8+k] = emb[v][l+k]  (B matrix for conv GEMM, N=128 pad, K=224 pad)
__global__ void k_build_ebt(const float* __restrict__ emb, _Float16* __restrict__ Ebt) {
  int idx = blockIdx.x * blockDim.x + threadIdx.x;
  if (idx >= 128 * 224) return;
  int l = idx / 224, vk = idx % 224;
  int v = vk >> 3, k = vk & 7;
  float val = (l < LOUT && vk < VOC * 8) ? emb[v * EMB + l + k] : 0.f;
  Ebt[idx] = (_Float16)val;
}

// ---------------- GIN pipeline ----------------
__global__ void k_init_agg78(const float* __restrict__ x, float* __restrict__ agg) {
  int idx = blockIdx.x * blockDim.x + threadIdx.x;
  if (idx >= N_NODES * 96) return;
  int node = idx / 96, c = idx % 96;
  agg[idx] = (c < 78) ? x[(size_t)node * 78 + c] : 0.f;
}

__global__ void k_edge_scatter78(const int* __restrict__ src, const int* __restrict__ dst,
                                 const float* __restrict__ x, float* __restrict__ agg) {
  int idx = blockIdx.x * blockDim.x + threadIdx.x;
  if (idx >= N_EDGES * 78) return;
  int e = idx / 78, f = idx % 78;
  atomicAdd(&agg[(size_t)dst[e] * 96 + f], x[(size_t)src[e] * 78 + f]);
}

__global__ void k_edge_scatter32(const int* __restrict__ src, const int* __restrict__ dst,
                                 const float* __restrict__ h, float* __restrict__ agg) {
  int idx = blockIdx.x * blockDim.x + threadIdx.x;
  if (idx >= N_EDGES * 32) return;
  int e = idx >> 5, f = idx & 31;
  atomicAdd(&agg[(size_t)dst[e] * 32 + f], h[(size_t)src[e] * 32 + f]);
}

// Fused GIN MLP: r = relu( relu(agg@W1+b1) @ W2 + b2 ), write r, accumulate BN stats.
// block = 128 threads = 4 waves; wave handles one 16-node tile (both 16-col halves).
__global__ void k_gin_mlp(const float* __restrict__ agg, int lda, int ksteps,
                          const _Float16* __restrict__ W1t, int ld1, const float* __restrict__ b1,
                          const _Float16* __restrict__ W2t, const float* __restrict__ b2,
                          float* __restrict__ hout, float* __restrict__ gstats) {
  __shared__ float tbuf[4][16][TLD];
  __shared__ float ssum[32], ssq[32];
  int tid = threadIdx.x, w = tid >> 5, lane = tid & 31;
  int l15 = lane & 15, hi = (lane >> 4) & 1;
  if (tid < 32) { ssum[tid] = 0.f; ssq[tid] = 0.f; }
  __syncthreads();

  int nb = (blockIdx.x * 4 + w) * 16;
  v8f acc0 = zero8(), acc1 = zero8();
  const float* arow = agg + (size_t)(nb + l15) * lda;
  for (int ks = 0; ks < ksteps; ++ks) {
    int kb = ks * 32;
    v16h a  = load_aF32(arow, kb, lane);
    v16h b0 = load_bT(W1t, ld1, 0, kb, lane);
    v16h bB = load_bT(W1t, ld1, 16, kb, lane);
    acc0 = wmma16(a, b0, acc0);
    acc1 = wmma16(a, bB, acc1);
  }
  float bi0 = b1[l15], bi1 = b1[16 + l15];
#pragma unroll
  for (int r = 0; r < 8; ++r) {
    int mrow = r + hi * 8;
    tbuf[w][mrow][l15]      = fmaxf(acc0[r] + bi0, 0.f);
    tbuf[w][mrow][16 + l15] = fmaxf(acc1[r] + bi1, 0.f);
  }
  __syncthreads();

  acc0 = zero8(); acc1 = zero8();
  {
    const float* trow = &tbuf[w][l15][0];
    int ko = hi ? 8 : 0;
    v16h a2;
#pragma unroll
    for (int i = 0; i < 8; ++i) { a2[i] = (_Float16)trow[ko + i]; a2[8 + i] = (_Float16)trow[ko + 16 + i]; }
    v16h b0 = load_bT(W2t, 32, 0, 0, lane);
    v16h bB = load_bT(W2t, 32, 16, 0, lane);
    acc0 = wmma16(a2, b0, acc0);
    acc1 = wmma16(a2, bB, acc1);
  }
  bi0 = b2[l15]; bi1 = b2[16 + l15];
  float s0 = 0.f, q0 = 0.f, s1 = 0.f, q1 = 0.f;
#pragma unroll
  for (int r = 0; r < 8; ++r) {
    int m = nb + r + hi * 8;
    float v0 = fmaxf(acc0[r] + bi0, 0.f);
    float v1 = fmaxf(acc1[r] + bi1, 0.f);
    hout[(size_t)m * 32 + l15]      = v0;
    hout[(size_t)m * 32 + 16 + l15] = v1;
    s0 += v0; q0 += v0 * v0; s1 += v1; q1 += v1 * v1;
  }
  atomicAdd(&ssum[l15], s0);      atomicAdd(&ssq[l15], q0);
  atomicAdd(&ssum[16 + l15], s1); atomicAdd(&ssq[16 + l15], q1);
  __syncthreads();
  if (tid < 32) { atomicAdd(&gstats[tid], ssum[tid]); atomicAdd(&gstats[32 + tid], ssq[tid]); }
}

__global__ void k_bn_final(const float* __restrict__ gstats, const float* __restrict__ g,
                           const float* __restrict__ be, float* __restrict__ bnp) {
  int f = threadIdx.x;
  if (f >= 32) return;
  const float inv = 1.f / (float)N_NODES;
  float mean = gstats[f] * inv;
  float var  = gstats[32 + f] * inv - mean * mean;
  float sc   = g[f] * rsqrtf(var + 1e-5f);
  bnp[f] = sc;
  bnp[32 + f] = be[f] - mean * sc;
}

__global__ void k_bn_apply(float* __restrict__ h, const float* __restrict__ bnp,
                           float* __restrict__ aggnext) {
  int idx = blockIdx.x * blockDim.x + threadIdx.x;
  if (idx >= N_NODES * 32) return;
  int f = idx & 31;
  float v = h[idx] * bnp[f] + bnp[32 + f];
  h[idx] = v;
  aggnext[idx] = v;  // self term for next layer's aggregation
}

// segment-sum pool over sorted batch; 16-node run accumulation to cut atomics 16x
__global__ void k_pool(const float* __restrict__ h, const int* __restrict__ batch,
                       float* __restrict__ pooled) {
  int idx = blockIdx.x * blockDim.x + threadIdx.x;
  int f = idx & 31, chunk = idx >> 5;
  if (chunk >= N_NODES / 16) return;
  int n0 = chunk * 16;
  float acc = 0.f;
  int bcur = batch[n0];
  for (int j = 0; j < 16; ++j) {
    int n = n0 + j;
    int bb = batch[n];
    if (bb != bcur) { atomicAdd(&pooled[bcur * 32 + f], acc); acc = 0.f; bcur = bb; }
    acc += h[(size_t)n * 32 + f];
  }
  atomicAdd(&pooled[bcur * 32 + f], acc);
}

__global__ void k_xd(const float* __restrict__ pooled, const float* __restrict__ W,
                     const float* __restrict__ bias, _Float16* __restrict__ xdxt) {
  int idx = blockIdx.x * blockDim.x + threadIdx.x;
  if (idx >= NB * 128) return;
  int b = idx >> 7, c = idx & 127;
  float a = bias[c];
  for (int k = 0; k < 32; ++k) a += pooled[b * 32 + k] * W[k * 128 + c];
  xdxt[(size_t)b * 256 + c] = (_Float16)fmaxf(a, 0.f);
}

// ---------------- protein branch: factorized conv, fused S-build + WMMA GEMM ----------------
// one block per batch element, 256 threads = 8 waves
__global__ void k_conv(const int* __restrict__ target, const float* __restrict__ convW,
                       const float* __restrict__ convb, const _Float16* __restrict__ Ebt,
                       _Float16* __restrict__ convA) {
  __shared__ float S[32][SLD];
  __shared__ int   tl[SEQ];
  int b = blockIdx.x, tid = threadIdx.x;
  for (int i = tid; i < 32 * SLD; i += 256) ((float*)S)[i] = 0.f;
  for (int i = tid; i < SEQ; i += 256)      tl[i] = target[(size_t)b * SEQ + i];
  __syncthreads();
  { // S[o][v*8+k] += W[o][i][k] for i with token v
    int o = tid >> 3, k = tid & 7;
    const float* Wp = convW + (size_t)o * (SEQ * 8) + k;
    for (int i = 0; i < SEQ; ++i) S[o][tl[i] * 8 + k] += Wp[i * 8];
  }
  __syncthreads();
  int w = tid >> 5, lane = tid & 31, l15 = lane & 15, hi = (lane >> 4) & 1;
  for (int t = w; t < 16; t += 8) {           // 2 m-tiles x 8 n-tiles
    int mi = t & 1, ni = t >> 1;
    v8f acc = zero8();
    const float* arow = &S[mi * 16 + l15][0];
    for (int ks = 0; ks < 7; ++ks) {          // K = 224 (208 real, zero-padded)
      int kb = ks * 32;
      v16h a  = load_aF32(arow, kb, lane);
      v16h bb = load_bT(Ebt, 224, ni * 16, kb, lane);
      acc = wmma16(a, bb, acc);
    }
#pragma unroll
    for (int r = 0; r < 8; ++r) {
      int o = mi * 16 + r + hi * 8;
      int l = ni * 16 + l15;
      float cb = convb[o];
      if (l < LOUT) convA[(size_t)b * CONV_K + o * LOUT + l] = (_Float16)(acc[r] + cb);
    }
  }
}

// ---------------- generic f16 WMMA GEMM: C = act(A @ Bt^T + bias) ----------------
// block = 128 threads = 4 waves; wave per 16x16 tile
__global__ void k_gemm_f16(const _Float16* __restrict__ A, int lda,
                           const _Float16* __restrict__ Bt, int ldb,
                           const float* __restrict__ bias,
                           _Float16* __restrict__ C, int ldc, int coff,
                           int mtiles, int ksteps, int relu) {
  int tid = threadIdx.x, w = tid >> 5, lane = tid & 31;
  int t = blockIdx.x * 4 + w;
  int mi = t % mtiles, ni = t / mtiles;
  int l15 = lane & 15, hi = (lane >> 4) & 1;
  v8f acc = zero8();
  for (int ks = 0; ks < ksteps; ++ks) {
    int kb = ks * 32;
    if (ks + 2 < ksteps) {
      __builtin_prefetch(A + (size_t)(mi * 16 + l15) * lda + kb + 64, 0, 1);
      __builtin_prefetch(Bt + (size_t)(ni * 16 + l15) * ldb + kb + 64, 0, 1);
    }
    v16h a = load_aF16(A, lda, mi * 16, kb, lane);
    v16h b = load_bT(Bt, ldb, ni * 16, kb, lane);
    acc = wmma16(a, b, acc);
  }
  float bv = bias[ni * 16 + l15];
#pragma unroll
  for (int r = 0; r < 8; ++r) {
    float v = acc[r] + bv;
    if (relu) v = fmaxf(v, 0.f);
    C[(size_t)(mi * 16 + r + hi * 8) * ldc + coff + ni * 16 + l15] = (_Float16)v;
  }
}

__global__ void k_out(const _Float16* __restrict__ xc2, const float* __restrict__ oW,
                      const float* __restrict__ ob, float* __restrict__ out) {
  int b = blockIdx.x * blockDim.x + threadIdx.x;
  if (b >= NB) return;
  float a = ob[0];
  for (int j = 0; j < 256; ++j) a += (float)xc2[(size_t)b * 256 + j] * oW[j];
  out[b] = a;
}

// ---------------- host orchestration ----------------
static inline int cdiv(long long a, long long b) { return (int)((a + b - 1) / b); }

extern "C" void kernel_launch(void* const* d_in, const int* in_sizes, int n_in,
                              void* d_out, int out_size, void* d_ws, size_t ws_size,
                              hipStream_t stream) {
  const float* x       = (const float*)d_in[0];
  const int*   ei      = (const int*)d_in[1];
  const int*   batch   = (const int*)d_in[2];
  const int*   target  = (const int*)d_in[3];
  const float* W1a     = (const float*)d_in[4];
  const float* b1a     = (const float*)d_in[5];
  const float* W2a     = (const float*)d_in[6];
  const float* b2a     = (const float*)d_in[7];
  const float* g1      = (const float*)d_in[8];
  const float* be1     = (const float*)d_in[9];
  const float* Ws1     = (const float*)d_in[10];
  const float* bs1     = (const float*)d_in[11];
  const float* Ws2     = (const float*)d_in[12];
  const float* bs2     = (const float*)d_in[13];
  const float* gs      = (const float*)d_in[14];
  const float* bes     = (const float*)d_in[15];
  const float* fc1xd_W = (const float*)d_in[16];
  const float* fc1xd_b = (const float*)d_in[17];
  const float* emb     = (const float*)d_in[18];
  const float* convW   = (const float*)d_in[19];
  const float* convb   = (const float*)d_in[20];
  const float* fcxt_W  = (const float*)d_in[21];
  const float* fcxt_b  = (const float*)d_in[22];
  const float* fc1_W   = (const float*)d_in[23];
  const float* fc1_b   = (const float*)d_in[24];
  const float* fc2_W   = (const float*)d_in[25];
  const float* fc2_b   = (const float*)d_in[26];
  const float* out_W   = (const float*)d_in[27];
  const float* out_b   = (const float*)d_in[28];
  float* out = (float*)d_out;

  const int* src = ei;
  const int* dst = ei + N_EDGES;

  // workspace layout (256B aligned)
  size_t off = 0;
  auto alloc = [&](size_t bytes) -> char* {
    char* p = (char*)d_ws + off;
    off = (off + bytes + 255) & ~(size_t)255;
    return p;
  };
  float*    agg96  = (float*)alloc((size_t)N_NODES * 96 * 4);
  float*    h      = (float*)alloc((size_t)N_NODES * 32 * 4);
  float*    agg32  = (float*)alloc((size_t)N_NODES * 32 * 4);
  float*    gstats = (float*)alloc(64 * 4);
  float*    bnp    = (float*)alloc(64 * 4);
  float*    pooled = (float*)alloc((size_t)NB * 32 * 4);
  _Float16* xdxt   = (_Float16*)alloc((size_t)NB * 256 * 2);
  _Float16* convA  = (_Float16*)alloc((size_t)NB * CONV_K * 2);
  _Float16* xc1    = (_Float16*)alloc((size_t)NB * 1024 * 2);
  _Float16* xc2    = (_Float16*)alloc((size_t)NB * 256 * 2);
  _Float16* Ebt    = (_Float16*)alloc(128 * 224 * 2);
  _Float16* W1aT   = (_Float16*)alloc(32 * 96 * 2);
  _Float16* W2aT   = (_Float16*)alloc(32 * 32 * 2);
  _Float16* WsT1   = (_Float16*)alloc(4 * 32 * 32 * 2);
  _Float16* WsT2   = (_Float16*)alloc(4 * 32 * 32 * 2);
  _Float16* fcxtT  = (_Float16*)alloc((size_t)128 * CONV_K * 2);
  _Float16* fc1T   = (_Float16*)alloc((size_t)1024 * 256 * 2);
  _Float16* fc2T   = (_Float16*)alloc((size_t)256 * 1024 * 2);
  (void)ws_size; (void)n_in; (void)in_sizes; (void)out_size;

  const int TB = 256;

  // --- weight prep (independent; front-loaded) ---
  k_transpose_f16<<<cdiv(32 * 96, TB), TB, 0, stream>>>(W1a, W1aT, 78, 32, 96);
  k_transpose_f16<<<cdiv(32 * 32, TB), TB, 0, stream>>>(W2a, W2aT, 32, 32, 32);
  for (int i = 0; i < 4; ++i) {
    k_transpose_f16<<<cdiv(32 * 32, TB), TB, 0, stream>>>(Ws1 + i * 1024, WsT1 + i * 1024, 32, 32, 32);
    k_transpose_f16<<<cdiv(32 * 32, TB), TB, 0, stream>>>(Ws2 + i * 1024, WsT2 + i * 1024, 32, 32, 32);
  }
  k_transpose_f16<<<cdiv(128 * CONV_K, TB), TB, 0, stream>>>(fcxt_W, fcxtT, CONV_K, 128, CONV_K);
  k_transpose_f16<<<cdiv(1024 * 256, TB), TB, 0, stream>>>(fc1_W, fc1T, 256, 1024, 256);
  k_transpose_f16<<<cdiv(256 * 1024, TB), TB, 0, stream>>>(fc2_W, fc2T, 1024, 256, 1024);
  k_build_ebt<<<cdiv(128 * 224, TB), TB, 0, stream>>>(emb, Ebt);

  // --- GIN layer 1 (78 -> 32 -> 32) ---
  k_init_agg78<<<cdiv((long long)N_NODES * 96, TB), TB, 0, stream>>>(x, agg96);
  k_edge_scatter78<<<cdiv((long long)N_EDGES * 78, TB), TB, 0, stream>>>(src, dst, x, agg96);
  hipMemsetAsync(gstats, 0, 64 * 4, stream);
  k_gin_mlp<<<N_NODES / 16 / 4, 128, 0, stream>>>(agg96, 96, 3, W1aT, 96, b1a, W2aT, b2a, h, gstats);
  k_bn_final<<<1, 32, 0, stream>>>(gstats, g1, be1, bnp);
  k_bn_apply<<<cdiv((long long)N_NODES * 32, TB), TB, 0, stream>>>(h, bnp, agg32);

  // --- GIN layers 2-5 (32 -> 32 -> 32) ---
  for (int i = 0; i < 4; ++i) {
    k_edge_scatter32<<<cdiv((long long)N_EDGES * 32, TB), TB, 0, stream>>>(src, dst, h, agg32);
    hipMemsetAsync(gstats, 0, 64 * 4, stream);
    k_gin_mlp<<<N_NODES / 16 / 4, 128, 0, stream>>>(agg32, 32, 1, WsT1 + i * 1024, 32, bs1 + i * 32,
                                                    WsT2 + i * 1024, bs2 + i * 32, h, gstats);
    k_bn_final<<<1, 32, 0, stream>>>(gstats, gs + i * 32, bes + i * 32, bnp);
    k_bn_apply<<<cdiv((long long)N_NODES * 32, TB), TB, 0, stream>>>(h, bnp, agg32);
  }

  // --- pool + drug head ---
  hipMemsetAsync(pooled, 0, (size_t)NB * 32 * 4, stream);
  k_pool<<<cdiv((long long)(N_NODES / 16) * 32, TB), TB, 0, stream>>>(h, batch, pooled);
  k_xd<<<cdiv(NB * 128, TB), TB, 0, stream>>>(pooled, fc1xd_W, fc1xd_b, xdxt);

  // --- protein branch: factorized conv + fcxt GEMM ---
  k_conv<<<NB, 256, 0, stream>>>(target, convW, convb, Ebt, convA);
  // xt = convA[1024x3872] @ fcxtT -> xdxt cols 128..255
  k_gemm_f16<<<(64 * 8) / 4, 128, 0, stream>>>(convA, CONV_K, fcxtT, CONV_K, fcxt_b,
                                               xdxt, 256, 128, 64, CONV_K / 32, 0);
  // --- joint head ---
  k_gemm_f16<<<(64 * 64) / 4, 128, 0, stream>>>(xdxt, 256, fc1T, 256, fc1_b,
                                                xc1, 1024, 0, 64, 256 / 32, 1);
  k_gemm_f16<<<(64 * 16) / 4, 128, 0, stream>>>(xc1, 1024, fc2T, 1024, fc2_b,
                                                xc2, 256, 0, 64, 1024 / 32, 1);
  k_out<<<cdiv(NB, TB), TB, 0, stream>>>(xc2, out_W, out_b, out);
}
